// AMIPRouterInference_14559939133632
// MI455X (gfx1250) — matrix operand order repaired and checked by reading, not compiled, not run on hardware
//
#include <hip/hip_runtime.h>
#include <math.h>

// Problem constants (from reference setup)
#define LL 2048   // seq len
#define DD 4096   // d_model
#define MM 1024   // masks (even rows)
#define UU 1024   // anchors (odd rows)
#define DPJ 512   // q/k proj dim
#define HH 2048   // expert hidden
#define KE 8      // experts

typedef __bf16 bf16;
typedef bf16  bf16x2 __attribute__((ext_vector_type(2)));
typedef bf16  bf16x8 __attribute__((ext_vector_type(8)));
typedef bf16  v16bf  __attribute__((ext_vector_type(16)));
typedef float v8f    __attribute__((ext_vector_type(8)));
typedef unsigned int u32x4 __attribute__((ext_vector_type(4)));
typedef int i32x4 __attribute__((ext_vector_type(4)));
typedef int i32x8 __attribute__((ext_vector_type(8)));

#if defined(__has_builtin)
#if __has_builtin(__builtin_amdgcn_tensor_load_to_lds) && __has_builtin(__builtin_amdgcn_s_wait_tensorcnt)
#define HAVE_TDM 1
#endif
#endif
#ifndef HAVE_TDM
#define HAVE_TDM 0
#endif

__device__ __forceinline__ bf16 f2bf(float f) { return (bf16)f; }  // v_cvt (RNE)

__global__ void kzero(float* __restrict__ p, int n) {
  int i = blockIdx.x * 256 + threadIdx.x;
  if (i < n) p[i] = 0.0f;
}

__global__ void kcvt(const float* __restrict__ h, bf16* __restrict__ hb, int n) {
  int i = blockIdx.x * 256 + threadIdx.x;
  if (i < n) hb[i] = f2bf(h[i]);
}

// gate[i][k] = softmax_k(h[2i] . Wr[:,k] + br[k]);  one block per mask
__global__ __launch_bounds__(256) void kgate(const float* __restrict__ h,
                                             const float* __restrict__ Wr,
                                             const float* __restrict__ br,
                                             float* __restrict__ gate) {
  __shared__ float red[256];
  __shared__ float logits[KE];
  const int i = blockIdx.x, tid = threadIdx.x;
  const float* hrow = h + (size_t)(2 * i) * DD;
  float acc[KE];
#pragma unroll
  for (int k = 0; k < KE; ++k) acc[k] = 0.0f;
  for (int d = tid; d < DD; d += 256) {
    float hv = hrow[d];
    const float* wr = Wr + (size_t)d * KE;
#pragma unroll
    for (int k = 0; k < KE; ++k) acc[k] += hv * wr[k];
  }
  for (int k = 0; k < KE; ++k) {
    red[tid] = acc[k];
    __syncthreads();
    for (int o = 128; o > 0; o >>= 1) {
      if (tid < o) red[tid] += red[tid + o];
      __syncthreads();
    }
    if (tid == 0) logits[k] = red[0] + br[k];
    __syncthreads();
  }
  if (tid == 0) {
    float mx = logits[0];
#pragma unroll
    for (int k = 1; k < KE; ++k) mx = fmaxf(mx, logits[k]);
    float ex[KE], s = 0.0f;
#pragma unroll
    for (int k = 0; k < KE; ++k) { ex[k] = expf(logits[k] - mx); s += ex[k]; }
    float inv = 1.0f / s;
#pragma unroll
    for (int k = 0; k < KE; ++k) gate[i * KE + k] = ex[k] * inv;
  }
}

// combine[i][s] over anchor window u = i-5+s, s in [0,10); softmax of q.k/sqrt(DP)
__global__ __launch_bounds__(320) void kcomb(const float* __restrict__ q,
                                             const float* __restrict__ kkv,
                                             float* __restrict__ comb) {
  __shared__ float sc[10];
  const int i = blockIdx.x, tid = threadIdx.x;
  const int w = tid >> 5, lane = tid & 31;
  const int u = i - 5 + w;
  const bool valid = (u >= 0) && (u < UU);
  float p = 0.0f;
  if (valid) {
    const float* qr = q + (size_t)i * DPJ;
    const float* kr = kkv + (size_t)u * DPJ;
    for (int d = lane; d < DPJ; d += 32) p += qr[d] * kr[d];
#pragma unroll
    for (int o = 16; o > 0; o >>= 1) p += __shfl_down(p, o);
  }
  if (lane == 0) sc[w] = valid ? (p * 0.044194173824159216f /*1/sqrt(512)*/) : -1e9f;
  __syncthreads();
  if (tid == 0) {
    float mx = -1e9f;
#pragma unroll
    for (int s = 0; s < 10; ++s) mx = fmaxf(mx, sc[s]);
    float ex[10], sum = 0.0f;
#pragma unroll
    for (int s = 0; s < 10; ++s) {
      ex[s] = (sc[s] > -1e8f) ? expf(sc[s] - mx) : 0.0f;
      sum += ex[s];
    }
    float inv = 1.0f / fmaxf(sum, 1e-8f);
#pragma unroll
    for (int s = 0; s < 10; ++s) comb[i * 10 + s] = ex[s] * inv;
  }
}

// ---------------- WMMA GEMM (bf16 in, f32 accumulate) ----------------
// MODE 0: S1[2048,2048] = [hb_anchor ; hb_mask] @ W1e(split halves), K=4096
// MODE 1: d_out[2i] += gate[i,e] * (Hbar @ W2e + bias), K=2048
// MODE 2: q  = h_masks   @ Wq + bias   (ND=512)
// MODE 3: k  = h_anchors @ Wk + bias   (ND=512)
#define LDST 40  // padded LDS row stride (elements); 80B keeps 16B alignment

__device__ __forceinline__ v16bf fragA(const bf16* p) {
  bf16x8 lo = *(const bf16x8*)p;          // K base .. +8
  bf16x8 hi = *(const bf16x8*)(p + 16);   // K base+16 .. +24
  return __builtin_shufflevector(lo, hi, 0, 1, 2, 3, 4, 5, 6, 7, 8, 9, 10, 11, 12, 13, 14, 15);
}
__device__ __forceinline__ v16bf fragB(const bf16* p) {
  bf16x8 lo = *(const bf16x8*)p;          // K base .. +8
  bf16x8 hi = *(const bf16x8*)(p + 16);   // K base+16 .. +24
  return __builtin_shufflevector(lo, hi, 0, 1, 2, 3, 4, 5, 6, 7, 8, 9, 10, 11, 12, 13, 14, 15);
}

#if HAVE_TDM
// One TDM load of a [128 rows x 32 elem] bf16 tile into LDS, 64B data rows
// padded by 16B (pad_interval=3 -> 16 DWORDs, pad_amount=3 -> 4 DWORDs) so the
// LDS row stride is 80B == LDST elements. D# per CDNA5 ISA ch.8.
__device__ __forceinline__ void tdmLoadA(unsigned ldsOff, unsigned long long gaddr,
                                         unsigned strideElems) {
  u32x4 g0;
  g0[0] = 1u;                                              // count=1 (valid)
  g0[1] = ldsOff;                                          // lds_addr
  g0[2] = (unsigned)(gaddr & 0xffffffffull);               // global_addr[31:0]
  g0[3] = (unsigned)((gaddr >> 32) & 0x1ffffffull) | (2u << 30);  // addr[56:32]|type=2
  i32x8 g1;
  g1[0] = (int)((1u << 16) | (1u << 20) | (3u << 22) | (3u << 25));
  g1[1] = (int)(32u << 16);                // tensor_dim0[15:0] @ bits 63:48
  g1[2] = (int)(128u << 16);               // tensor_dim0 hi=0 | tensor_dim1[15:0]
  g1[3] = (int)(32u << 16);                // tensor_dim1 hi=0 | tile_dim0=32
  g1[4] = (int)(128u);                     // tile_dim1=128 | tile_dim2=0
  g1[5] = (int)strideElems;                // tensor_dim0_stride[31:0]
  g1[6] = 0;
  g1[7] = 0;
  i32x4 z4 = {0, 0, 0, 0};
#if __clang_major__ >= 23
  i32x8 z8 = {0, 0, 0, 0, 0, 0, 0, 0};
  __builtin_amdgcn_tensor_load_to_lds(g0, g1, z4, z4, z8, 0);
#else
  __builtin_amdgcn_tensor_load_to_lds(g0, g1, z4, z4, 0);
#endif
}
#endif

template <int MODE>
__global__ __launch_bounds__(256) void kgemm(const bf16* __restrict__ Abase,
                                             const float* __restrict__ Bsrc,
                                             float* __restrict__ Out,
                                             const float* __restrict__ gate,
                                             const float* __restrict__ bias,
                                             float* __restrict__ outp,
                                             int e) {
  constexpr int KD = (MODE == 1) ? HH : DD;                       // reduction dim
  constexpr int ND = (MODE == 0) ? HH : ((MODE == 1) ? DD : DPJ); // output cols
  constexpr int LDA = (MODE == 1) ? HH : DD;                      // A row stride
  constexpr int NT = KD / 32;

  __shared__ bf16 lA[2][128 * LDST];
  __shared__ bf16 lB[2][128 * LDST];

  const int tid = threadIdx.x;
  const int nb = blockIdx.x, mb = blockIdx.y;
  const int wbase = (MODE == 0 && mb >= 8) ? DD : 0;  // W1 row half select
  const int nbase = nb * 128;

  const int lane = tid & 31, wid = tid >> 5;
  const int wr = wid >> 1, wc = wid & 1;
  const int kh = lane >> 4, l15 = lane & 15;

  // B staging geometry: each thread covers 2 k-rows x 4 n-cols, twice
  int k2_[2], n4_[2];
#pragma unroll
  for (int it = 0; it < 2; ++it) {
    const int u4 = tid + it * 256;
    k2_[it] = (u4 >> 5) << 1;
    n4_[it] = (u4 & 31) << 2;
  }
  float4 gw[2][2];

#if HAVE_TDM
  const int R0 = mb * 128;
  size_t arow0;
  if (MODE == 0) arow0 = (R0 < MM) ? (size_t)(2 * R0 + 1) : (size_t)(2 * (R0 - MM));
  else if (MODE == 1) arow0 = (size_t)R0;
  else if (MODE == 2) arow0 = (size_t)(2 * R0);
  else arow0 = (size_t)(2 * R0 + 1);
  const unsigned astride = (MODE == 1) ? (unsigned)LDA : (unsigned)(2 * LDA);
  const unsigned long long abase =
      (unsigned long long)(uintptr_t)(Abase + arow0 * LDA);
  const unsigned ldsA0 = (unsigned)(uintptr_t)(&lA[0][0]);
  const unsigned ldsA1 = (unsigned)(uintptr_t)(&lA[1][0]);
#else
  const int R = mb * 128 + (tid >> 1);
  size_t grow;
  if (MODE == 0) grow = (R < MM) ? (size_t)(2 * R + 1) : (size_t)(2 * (R - MM));
  else if (MODE == 1) grow = (size_t)R;
  else if (MODE == 2) grow = (size_t)(2 * R);
  else grow = (size_t)(2 * R + 1);
  const bf16* aSrc = Abase + grow * LDA + (size_t)((tid & 1) * 16);
  bf16x8 ga[2];
#endif

  v8f acc[2][4];
  const v8f vz = {0.f, 0.f, 0.f, 0.f, 0.f, 0.f, 0.f, 0.f};
#pragma unroll
  for (int mi = 0; mi < 2; ++mi)
#pragma unroll
    for (int ni = 0; ni < 4; ++ni) acc[mi][ni] = vz;

  auto loadG = [&](int kk) {  // global -> registers
#pragma unroll
    for (int it = 0; it < 2; ++it) {
      const float* bp = Bsrc + (size_t)(wbase + kk + k2_[it]) * ND + nbase + n4_[it];
      gw[it][0] = *(const float4*)(bp);
      gw[it][1] = *(const float4*)(bp + ND);
    }
#if !HAVE_TDM
    const bf16x8* s = (const bf16x8*)(aSrc + kk);
    ga[0] = s[0];
    ga[1] = s[1];
#endif
  };

  auto storeL = [&](int p) {  // registers -> LDS (hw cvt + bf16x2 stores)
    bf16* B = &lB[p][0];
#pragma unroll
    for (int it = 0; it < 2; ++it) {
      const int k2 = k2_[it], n4 = n4_[it];
      bf16x2 vx = {f2bf(gw[it][0].x), f2bf(gw[it][1].x)};
      bf16x2 vy = {f2bf(gw[it][0].y), f2bf(gw[it][1].y)};
      bf16x2 vz2 = {f2bf(gw[it][0].z), f2bf(gw[it][1].z)};
      bf16x2 vw = {f2bf(gw[it][0].w), f2bf(gw[it][1].w)};
      *(bf16x2*)&B[(n4 + 0) * LDST + k2] = vx;
      *(bf16x2*)&B[(n4 + 1) * LDST + k2] = vy;
      *(bf16x2*)&B[(n4 + 2) * LDST + k2] = vz2;
      *(bf16x2*)&B[(n4 + 3) * LDST + k2] = vw;
    }
#if !HAVE_TDM
    bf16* d = &lA[p][(tid >> 1) * LDST + (tid & 1) * 16];
    *(bf16x8*)d = ga[0];
    *(bf16x8*)(d + 8) = ga[1];
#endif
  };

  auto compute = [&](int p) {  // LDS fragments + 8 WMMAs per wave
    const bf16* LA = &lA[p][0];
    const bf16* LB = &lB[p][0];
    v16bf af[2], bfr[4];
#pragma unroll
    for (int mi = 0; mi < 2; ++mi)
      af[mi] = fragA(&LA[(wr * 32 + mi * 16 + l15) * LDST + kh * 8]);
#pragma unroll
    for (int ni = 0; ni < 4; ++ni)
      bfr[ni] = fragB(&LB[(wc * 64 + ni * 16 + l15) * LDST + kh * 8]);
#pragma unroll
    for (int mi = 0; mi < 2; ++mi)
#pragma unroll
      for (int ni = 0; ni < 4; ++ni)
        acc[mi][ni] = __builtin_amdgcn_wmma_f32_16x16x32_bf16(
            false, af[mi], false, bfr[ni], (short)0, acc[mi][ni], false, false);
  };

  // ---- software pipeline: one barrier per K-step, TDM/global overlap compute ----
  loadG(0);
#if HAVE_TDM
  if (wid == 0) tdmLoadA(ldsA0, abase, astride);
#endif
  storeL(0);
  for (int t = 0; t < NT; ++t) {
    const int p = t & 1;
    if (t + 1 < NT) loadG((t + 1) * 32);
#if HAVE_TDM
    if (wid == 0) __builtin_amdgcn_s_wait_tensorcnt(0);  // tile t's TDM done
#endif
    __syncthreads();  // tile t staged & visible; buf p^1 free for overwrite
#if HAVE_TDM
    if (t + 1 < NT && wid == 0)
      tdmLoadA(p ? ldsA0 : ldsA1, abase + (unsigned long long)(2 * (t + 1) * 32), astride);
#endif
    compute(p);
    if (t + 1 < NT) storeL(p ^ 1);
  }

  // epilogue; C layout: VGPR v -> row = base + v + 8*(lane>=16), col = base + (lane&15)
#pragma unroll
  for (int mi = 0; mi < 2; ++mi) {
#pragma unroll
    for (int ni = 0; ni < 4; ++ni) {
      const int orow0 = mb * 128 + wr * 32 + mi * 16 + kh * 8;
      const int ocol = nbase + wc * 64 + ni * 16 + l15;
      if (MODE == 0) {
#pragma unroll
        for (int v = 0; v < 8; ++v)
          Out[(size_t)(orow0 + v) * ND + ocol] = acc[mi][ni][v];
      } else if (MODE == 1) {
        const float bv = bias[ocol];
#pragma unroll
        for (int v = 0; v < 8; ++v) {
          const int i = orow0 + v;
          const float g = gate[i * KE + e];
          outp[(size_t)(2 * i) * DD + ocol] += g * (acc[mi][ni][v] + bv);
        }
      } else {  // MODE 2/3: q/k projections with bias
        const float bv = bias[ocol];
#pragma unroll
        for (int v = 0; v < 8; ++v)
          Out[(size_t)(orow0 + v) * ND + ocol] = acc[mi][ni][v] + bv;
      }
    }
  }
}

// Hbar[i] = sum_s combine[i,s] * gelu(S1[anchor u] + S1[mask i] + b1e)
__global__ __launch_bounds__(256) void khbar(const float* __restrict__ S1,
                                             const float* __restrict__ b1e,
                                             const float* __restrict__ comb,
                                             bf16* __restrict__ Hbar) {
  __shared__ float cw[10];
  const int i = blockIdx.x, tid = threadIdx.x;
  if (tid < 10) cw[tid] = comb[i * 10 + tid];
  __syncthreads();
  const float* sm = S1 + (size_t)(MM + i) * HH;
#pragma unroll
  for (int j = 0; j < HH / 256; ++j) {
    const int c = tid + j * 256;
    const float xm = sm[c] + b1e[c];
    float a = 0.0f;
#pragma unroll
    for (int s = 0; s < 10; ++s) {
      const float w = cw[s];
      if (w != 0.0f) {
        const int u = i - 5 + s;
        const float x = S1[(size_t)u * HH + c] + xm;
        a += w * 0.5f * x * (1.0f + erff(x * 0.70710678118654752f));
      }
    }
    Hbar[(size_t)i * HH + c] = f2bf(a);
  }
}

extern "C" void kernel_launch(void* const* d_in, const int* in_sizes, int n_in,
                              void* d_out, int out_size, void* d_ws, size_t ws_size,
                              hipStream_t stream) {
  (void)in_sizes; (void)n_in; (void)out_size; (void)ws_size;
  const float* h  = (const float*)d_in[0];
  const float* Wr = (const float*)d_in[4];
  const float* br = (const float*)d_in[5];
  const float* W1 = (const float*)d_in[6];
  const float* b1 = (const float*)d_in[7];
  const float* W2 = (const float*)d_in[8];
  const float* b2 = (const float*)d_in[9];
  const float* Wq = (const float*)d_in[10];
  const float* bq = (const float*)d_in[11];
  const float* Wk = (const float*)d_in[12];
  const float* bk = (const float*)d_in[13];
  float* out = (float*)d_out;

  char* ws = (char*)d_ws;
  size_t o = 0;
  bf16* hb = (bf16*)(ws + o);    o += (size_t)LL * DD * 2;      // 16.8 MB
  float* S1 = (float*)(ws + o);  o += (size_t)2 * MM * HH * 4;  // 16.8 MB
  bf16* Hb = (bf16*)(ws + o);    o += (size_t)MM * HH * 2;      //  4.2 MB
  float* q = (float*)(ws + o);   o += (size_t)MM * DPJ * 4;     //  2.1 MB
  float* kkv = (float*)(ws + o); o += (size_t)UU * DPJ * 4;     //  2.1 MB
  float* gate = (float*)(ws + o); o += (size_t)MM * KE * 4;
  float* comb = (float*)(ws + o); o += (size_t)MM * 10 * 4;

  const int nTot = LL * DD;
  kzero<<<(nTot + 255) / 256, 256, 0, stream>>>(out, nTot);
  kcvt<<<(nTot + 255) / 256, 256, 0, stream>>>(h, hb, nTot);
  kgate<<<MM, 256, 0, stream>>>(h, Wr, br, gate);
  // q/k projections on the WMMA path (MODE 2 / MODE 3)
  kgemm<2><<<dim3(DPJ / 128, MM / 128), 256, 0, stream>>>(hb, Wq, q, nullptr, bq, nullptr, 0);
  kgemm<3><<<dim3(DPJ / 128, UU / 128), 256, 0, stream>>>(hb, Wk, kkv, nullptr, bk, nullptr, 0);
  kcomb<<<MM, 320, 0, stream>>>(q, kkv, comb);

  for (int e = 0; e < KE; ++e) {
    const float* W1e = W1 + (size_t)e * 2 * DD * HH;
    const float* b1e = b1 + (size_t)e * HH;
    const float* W2e = W2 + (size_t)e * HH * DD;
    const float* b2e = b2 + (size_t)e * DD;
    kgemm<0><<<dim3(HH / 128, (2 * MM) / 128), 256, 0, stream>>>(hb, W1e, S1, nullptr, nullptr, nullptr, e);
    khbar<<<MM, 256, 0, stream>>>(S1, b1e, comb, Hb);
    kgemm<1><<<dim3(DD / 128, MM / 128), 256, 0, stream>>>(Hb, W2e, nullptr, gate, b2e, out, e);
  }
}
